// Transformer_pointdecoder_base_v1_63840393887957
// MI455X (gfx1250) — matrix operand order repaired and checked by reading, not compiled, not run on hardware
//
#include <hip/hip_runtime.h>
#include <math.h>

// ---- CDNA5 WMMA types --------------------------------------------------
typedef __attribute__((ext_vector_type(16))) __bf16 v16bf;
typedef __attribute__((ext_vector_type(8)))  __bf16 v8bf;
typedef __attribute__((ext_vector_type(8)))  float  v8f;
typedef unsigned short bf16_t;   // plumbing type (host-safe)

#define NB   8       // batch
#define LAT  1024
#define KD   128
#define MD   256
#define NN   4096
#define EPSI 1e-5f

__device__ __forceinline__ v16bf combine16(v8bf lo, v8bf hi) {
  v16bf r;
#pragma unroll
  for (int i = 0; i < 8; ++i) { r[i] = lo[i]; r[8 + i] = hi[i]; }
  return r;
}

__device__ __forceinline__ v8f wmma_bf16(v16bf a, v16bf b, v8f c) {
  // D(f32 16x16) = A(bf16 16x32) * B(bf16 32x16) + C
  return __builtin_amdgcn_wmma_f32_16x16x32_bf16(false, a, false, b, (short)0, c,
                                                 false, false);
}

// Issue 4 x 16B async global->LDS copies (one 32x256 bf16 K block / 256 thr)
__device__ __forceinline__ void stage_k_async(unsigned ldsBase,
                                              const __bf16* kbase, int tid) {
  const char* ksrc = (const char*)kbase;
#pragma unroll
  for (int u = 0; u < 4; ++u) {
    unsigned off = u * 4096u + (unsigned)tid * 16u;
    asm volatile("global_load_async_to_lds_b128 %0, %1, off"
                 :
                 : "v"(ldsBase + off), "v"(ksrc + off)
                 : "memory");
  }
}

// Gather 32 keys of one feature column into 4 packed v8bf registers
__device__ __forceinline__ void gather_v(const __bf16* vbase, int m,
                                         v8bf pk[4]) {
  const __bf16* vcol = vbase + m;
#pragma unroll
  for (int jj = 0; jj < 4; ++jj)
#pragma unroll
    for (int q = 0; q < 8; ++q) pk[jj][q] = vcol[(jj * 8 + q) * MD];
}

__device__ __forceinline__ void store_v(__bf16* VtBuf, int m, const v8bf pk[4]) {
#pragma unroll
  for (int jj = 0; jj < 4; ++jj) *(v8bf*)&VtBuf[m * 32 + jj * 8] = pk[jj];
}

// ---- fc1: [8,1024] x [32768,1024]^T -> h [B][k=128][m=256] -------------
__global__ void __launch_bounds__(256) k_fc1(const float* __restrict__ x,
                                             const float* __restrict__ fw,
                                             const float* __restrict__ fb,
                                             float* __restrict__ h) {
  __shared__ float xs[NB * LAT];
  for (int idx = threadIdx.x; idx < NB * LAT; idx += 256) xs[idx] = x[idx];
  __syncthreads();
  const int wave = threadIdx.x >> 5, lane = threadIdx.x & 31;
  const int r = blockIdx.x * 8 + wave;            // fc1 output row (0..32767)
  float acc[NB];
#pragma unroll
  for (int b = 0; b < NB; ++b) acc[b] = 0.f;
  const float* wrow = fw + (size_t)r * LAT;
  for (int i = 0; i < LAT / 32; ++i) {
    float wv = wrow[i * 32 + lane];
#pragma unroll
    for (int b = 0; b < NB; ++b) acc[b] += wv * xs[b * LAT + i * 32 + lane];
  }
#pragma unroll
  for (int msk = 16; msk >= 1; msk >>= 1)
#pragma unroll
    for (int b = 0; b < NB; ++b) acc[b] += __shfl_xor(acc[b], msk, 32);
  if (lane == 0) {
    float bias = fb[r];
    int m = r >> 7, k = r & 127;                  // row = m*128 + k
#pragma unroll
    for (int b = 0; b < NB; ++b) h[(b * KD + k) * MD + m] = acc[b] + bias;
  }
}

// ---- layer1: conv1(128->64) + BN + relu (q/k share conv, v separate) ---
__global__ void __launch_bounds__(256) k_layer1(
    const float* __restrict__ h, const float* __restrict__ w1q,
    const float* __restrict__ b1q, const float* __restrict__ w1v,
    const float* __restrict__ b1v,
    const float* gq, const float* beq, const float* mq, const float* vq,
    const float* gk, const float* bek, const float* mk, const float* vk,
    const float* gv, const float* bev, const float* mv, const float* vv,
    float* __restrict__ Q1, float* __restrict__ K1, float* __restrict__ V1) {
  int gid = blockIdx.x * 256 + threadIdx.x;       // < 8*64*256
  int m = gid & 255, o = (gid >> 8) & 63, b = gid >> 14;
  float aq = b1q[o], av = b1v[o];
  const float* hb = h + b * KD * MD + m;
#pragma unroll 4
  for (int k = 0; k < KD; ++k) {
    float hv = hb[k * MD];
    aq += w1q[o * KD + k] * hv;
    av += w1v[o * KD + k] * hv;
  }
  float sq = gq[o] * rsqrtf(vq[o] + EPSI), tq = beq[o] - mq[o] * sq;
  float sk = gk[o] * rsqrtf(vk[o] + EPSI), tk = bek[o] - mk[o] * sk;
  float sv = gv[o] * rsqrtf(vv[o] + EPSI), tv = bev[o] - mv[o] * sv;
  Q1[gid] = fmaxf(aq * sq + tq, 0.f);
  K1[gid] = fmaxf(aq * sk + tk, 0.f);
  V1[gid] = fmaxf(av * sv + tv, 0.f);
}

// ---- layer2: conv2(64->128)+BN+relu, store bf16 TRANSPOSED [b][m][c] ---
__global__ void __launch_bounds__(256) k_layer2(
    const float* __restrict__ Q1, const float* __restrict__ K1,
    const float* __restrict__ V1, const float* __restrict__ w2q,
    const float* __restrict__ b2q, const float* __restrict__ w2v,
    const float* __restrict__ b2v,
    const float* gq, const float* beq, const float* mq, const float* vq,
    const float* gk, const float* bek, const float* mk, const float* vk,
    const float* gv, const float* bev, const float* mv, const float* vv,
    bf16_t* __restrict__ Q2t_, bf16_t* __restrict__ K2t_,
    bf16_t* __restrict__ V2t_) {
  __bf16* Q2t = reinterpret_cast<__bf16*>(Q2t_);
  __bf16* K2t = reinterpret_cast<__bf16*>(K2t_);
  __bf16* V2t = reinterpret_cast<__bf16*>(V2t_);
  int gid = blockIdx.x * 256 + threadIdx.x;       // < 8*128*256
  int m = gid & 255, o = (gid >> 8) & 127, b = gid >> 15;
  float aq = b2q[o], ak = b2q[o], av = b2v[o];
  const float* q1b = Q1 + (b * 64) * MD + m;
  const float* k1b = K1 + (b * 64) * MD + m;
  const float* v1b = V1 + (b * 64) * MD + m;
#pragma unroll 4
  for (int k = 0; k < 64; ++k) {
    float wq = w2q[o * 64 + k], wv = w2v[o * 64 + k];
    aq += wq * q1b[k * MD];
    ak += wq * k1b[k * MD];
    av += wv * v1b[k * MD];
  }
  float sq = gq[o] * rsqrtf(vq[o] + EPSI), tq = beq[o] - mq[o] * sq;
  float sk = gk[o] * rsqrtf(vk[o] + EPSI), tk = bek[o] - mk[o] * sk;
  float sv = gv[o] * rsqrtf(vv[o] + EPSI), tv = bev[o] - mv[o] * sv;
  int tidx = (b * MD + m) * KD + o;               // [b][m][c]
  Q2t[tidx] = (__bf16)fmaxf(aq * sq + tq, 0.f);
  K2t[tidx] = (__bf16)fmaxf(ak * sk + tk, 0.f);
  V2t[tidx] = (__bf16)fmaxf(av * sv + tv, 0.f);
}

// ---- convert c3 weights fp32 -> bf16 -----------------------------------
__global__ void __launch_bounds__(256) k_cvtw(
    const float* __restrict__ wq, const float* __restrict__ wk,
    const float* __restrict__ wv, bf16_t* __restrict__ oq_,
    bf16_t* __restrict__ ok_, bf16_t* __restrict__ ov_) {
  __bf16* oq = reinterpret_cast<__bf16*>(oq_);
  __bf16* ok = reinterpret_cast<__bf16*>(ok_);
  __bf16* ov = reinterpret_cast<__bf16*>(ov_);
  int gid = blockIdx.x * 256 + threadIdx.x;       // < 4096*128
  oq[gid] = (__bf16)wq[gid];
  ok[gid] = (__bf16)wk[gid];
  ov[gid] = (__bf16)wv[gid];
}

// ---- layer3 GEMM via WMMA: Out[b][n][m] = W3[n,:]·Xt[b][m][:] + bias ---
// grid (32, 8), block 256 (8 waves, 16 n-rows each). 64 wmma per wave.
__global__ void __launch_bounds__(256) k_layer3(
    const bf16_t* __restrict__ W3_, const float* __restrict__ b3,
    const bf16_t* __restrict__ Xt_, bf16_t* __restrict__ Out_, float scale) {
  const __bf16* W3 = reinterpret_cast<const __bf16*>(W3_);
  const __bf16* Xt = reinterpret_cast<const __bf16*>(Xt_);
  __bf16* Out = reinterpret_cast<__bf16*>(Out_);
  const int b = blockIdx.y;
  const int wave = threadIdx.x >> 5, lane = threadIdx.x & 31;
  const int g = lane >> 4, ln = lane & 15;
  const int n0 = blockIdx.x * 128 + wave * 16;
  // A fragments: W3 rows (M=n, K=c), two contiguous 8-elem runs per lane
  v16bf Af[4];
  const __bf16* wrow = W3 + (n0 + ln) * KD;
#pragma unroll
  for (int ck = 0; ck < 4; ++ck) {
    v8bf lo = *(const v8bf*)(wrow + ck * 32 + g * 16);
    v8bf hi = *(const v8bf*)(wrow + ck * 32 + g * 16 + 8);
    Af[ck] = combine16(lo, hi);
  }
  float biasv[8];
#pragma unroll
  for (int r = 0; r < 8; ++r) biasv[r] = b3[n0 + g * 8 + r];
#pragma unroll
  for (int mt = 0; mt < 16; ++mt) {
    v8f acc = {};
#pragma unroll
    for (int ck = 0; ck < 4; ++ck) {
      // B element (K=c, N=m): Xt row m, contiguous 16 c values
      v16bf Bf = *(const v16bf*)(Xt + (b * MD + mt * 16 + ln) * KD + ck * 32 + g * 16);
      acc = wmma_bf16(Af[ck], Bf, acc);
    }
#pragma unroll
    for (int r = 0; r < 8; ++r) {
      float f = (acc[r] + biasv[r]) * scale;
      Out[(b * NN + n0 + g * 8 + r) * MD + mt * 16 + ln] = (__bf16)f;
    }
  }
}

// ---- flash attention + output projection (double-buffered staging) -----
// grid (32, 8), block 256. Wave owns 16 query rows, O = 16x256 f32 accum.
__global__ void __launch_bounds__(256) k_attn(
    const bf16_t* __restrict__ Qb_, const bf16_t* __restrict__ Kb_,
    const bf16_t* __restrict__ Vb_, const float* __restrict__ co_w,
    const float* __restrict__ co_b, float* __restrict__ out) {
  const __bf16* Qb = reinterpret_cast<const __bf16*>(Qb_);
  const __bf16* Kb = reinterpret_cast<const __bf16*>(Kb_);
  const __bf16* Vb = reinterpret_cast<const __bf16*>(Vb_);
  __shared__ __bf16 Kl[2][32 * 256];  // K block ping-pong, row-major [j][m]
  __shared__ __bf16 Vt[2][256 * 32];  // V block ping-pong, transposed [m][j]
  __shared__ __bf16 Ps[8][16 * 32];   // per-wave P scratch [i][j]
  const int tid = threadIdx.x;
  const int b = blockIdx.y;
  const int wave = tid >> 5, lane = tid & 31;
  const int g = lane >> 4, ln = lane & 15;
  const int iw = blockIdx.x * 128 + wave * 16;
  const v8f vzero = {};
  // Q fragments: 8 chunks of K(feature)=32, resident for whole kernel
  v16bf Qf[8];
  const __bf16* qrow = Qb + (size_t)(b * NN + iw + ln) * MD;
#pragma unroll
  for (int ck = 0; ck < 8; ++ck) {
    v8bf lo = *(const v8bf*)(qrow + ck * 32 + g * 16);
    v8bf hi = *(const v8bf*)(qrow + ck * 32 + g * 16 + 8);
    Qf[ck] = combine16(lo, hi);
  }
  v8f O[16];
#pragma unroll
  for (int t = 0; t < 16; ++t) O[t] = vzero;
  float m_run[8], l_run[8];
#pragma unroll
  for (int r = 0; r < 8; ++r) { m_run[r] = -1e30f; l_run[r] = 0.f; }

  // LDS byte offsets (generic address truncated to 32 bits == LDS offset)
  const unsigned ldsK[2] = {(unsigned)(size_t)(&Kl[0][0]),
                            (unsigned)(size_t)(&Kl[1][0])};
  const __bf16* kvbase = Kb + (size_t)b * NN * MD;
  const __bf16* vvbase = Vb + (size_t)b * NN * MD;

  // ---- prologue: stage chunk 0 into buffer 0 ----
  {
    stage_k_async(ldsK[0], kvbase, tid);
    v8bf pk[4];
    gather_v(vvbase, tid, pk);
    store_v(&Vt[0][0], tid, pk);
    asm volatile("s_wait_asynccnt 0x0" ::: "memory");
  }
  __syncthreads();

  int buf = 0;
  for (int j0 = 0; j0 < NN; j0 += 32) {
    const int jn = j0 + 32;
    const bool hasNext = jn < NN;
    // ---- kick off next chunk's staging into the other buffer ----
    v8bf pk[4];
    if (hasNext) {
      stage_k_async(ldsK[buf ^ 1], kvbase + (size_t)jn * MD, tid);
      gather_v(vvbase + (size_t)jn * MD, tid, pk);   // loads in flight
    }
    const __bf16* Klb = &Kl[buf][0];
    const __bf16* Vtb = &Vt[buf][0];
    // ---- S = Q·K^T for 16 rows x 32 keys (two 16x16 tiles) ----
    v8f S0 = vzero, S1 = vzero;
#pragma unroll
    for (int ck = 0; ck < 8; ++ck) {
      v16bf B0 = *(const v16bf*)&Klb[ln * 256 + ck * 32 + g * 16];
      S0 = wmma_bf16(Qf[ck], B0, S0);
      v16bf B1 = *(const v16bf*)&Klb[(16 + ln) * 256 + ck * 32 + g * 16];
      S1 = wmma_bf16(Qf[ck], B1, S1);
    }
    // ---- online softmax (row = r + g*8, cols live in a 16-lane half) ----
    float tmax[8];
#pragma unroll
    for (int r = 0; r < 8; ++r) tmax[r] = fmaxf(S0[r], S1[r]);
#pragma unroll
    for (int msk = 8; msk >= 1; msk >>= 1)
#pragma unroll
      for (int r = 0; r < 8; ++r)
        tmax[r] = fmaxf(tmax[r], __shfl_xor(tmax[r], msk, 32));
    float corr[8];
#pragma unroll
    for (int r = 0; r < 8; ++r) {
      float mn = fmaxf(m_run[r], tmax[r]);
      corr[r] = __expf(m_run[r] - mn);
      m_run[r] = mn;
    }
    float rs[8];
#pragma unroll
    for (int r = 0; r < 8; ++r) {
      S0[r] = __expf(S0[r] - m_run[r]);
      S1[r] = __expf(S1[r] - m_run[r]);
      rs[r] = S0[r] + S1[r];
    }
#pragma unroll
    for (int msk = 8; msk >= 1; msk >>= 1)
#pragma unroll
      for (int r = 0; r < 8; ++r) rs[r] += __shfl_xor(rs[r], msk, 32);
#pragma unroll
    for (int r = 0; r < 8; ++r) l_run[r] = l_run[r] * corr[r] + rs[r];
#pragma unroll
    for (int t = 0; t < 16; ++t)
#pragma unroll
      for (int r = 0; r < 8; ++r) O[t][r] *= corr[r];
    // P (C layout) -> LDS -> A-fragment layout (wave-private, in-order DS)
    __bf16* ps = &Ps[wave][0];
#pragma unroll
    for (int r = 0; r < 8; ++r) {
      ps[(r + g * 8) * 32 + ln] = (__bf16)S0[r];
      ps[(r + g * 8) * 32 + 16 + ln] = (__bf16)S1[r];
    }
    v8bf plo = *(const v8bf*)&ps[ln * 32 + g * 16];
    v8bf phi = *(const v8bf*)&ps[ln * 32 + g * 16 + 8];
    v16bf Pf = combine16(plo, phi);
    // ---- O += P(16x32) * V(32x256) ----
#pragma unroll
    for (int mt = 0; mt < 16; ++mt) {
      v16bf Vf = *(const v16bf*)&Vtb[(mt * 16 + ln) * 32 + g * 16];
      O[mt] = wmma_bf16(Pf, Vf, O[mt]);
    }
    // ---- finish next chunk's staging, single barrier per chunk ----
    if (hasNext) store_v(&Vt[buf ^ 1][0], tid, pk);
    asm volatile("s_wait_asynccnt 0x0" ::: "memory");
    __syncthreads();
    buf ^= 1;
  }
  // normalize
  float inv[8];
#pragma unroll
  for (int r = 0; r < 8; ++r) inv[r] = 1.0f / l_run[r];
#pragma unroll
  for (int t = 0; t < 16; ++t)
#pragma unroll
    for (int r = 0; r < 8; ++r) O[t][r] *= inv[r];
  // output projection: out[b][i][o] = sum_m O[i][m]*co_w[o][m] + co_b[o]
  for (int o = 0; o < 3; ++o) {
    float acc[8];
#pragma unroll
    for (int r = 0; r < 8; ++r) acc[r] = 0.f;
#pragma unroll
    for (int mt = 0; mt < 16; ++mt) {
      float wv = co_w[o * MD + mt * 16 + ln];
#pragma unroll
      for (int r = 0; r < 8; ++r) acc[r] += O[mt][r] * wv;
    }
#pragma unroll
    for (int msk = 8; msk >= 1; msk >>= 1)
#pragma unroll
      for (int r = 0; r < 8; ++r) acc[r] += __shfl_xor(acc[r], msk, 32);
    if (ln == 0) {
      float cb = co_b[o];
#pragma unroll
      for (int r = 0; r < 8; ++r)
        out[(size_t)(b * NN + iw + r + g * 8) * 3 + o] = acc[r] + cb;
    }
  }
}

// ---- host launch -------------------------------------------------------
extern "C" void kernel_launch(void* const* d_in, const int* in_sizes, int n_in,
                              void* d_out, int out_size, void* d_ws,
                              size_t ws_size, hipStream_t stream) {
  const float* x     = (const float*)d_in[0];
  const float* fc1_w = (const float*)d_in[1];
  const float* fc1_b = (const float*)d_in[2];
  const float* c1q_w = (const float*)d_in[3];
  const float* c1q_b = (const float*)d_in[4];
  const float* c2q_w = (const float*)d_in[5];
  const float* c2q_b = (const float*)d_in[6];
  const float* c3q_w = (const float*)d_in[7];
  const float* c3q_b = (const float*)d_in[8];
  const float* c3k_w = (const float*)d_in[9];
  const float* c3k_b = (const float*)d_in[10];
  const float* c1v_w = (const float*)d_in[11];
  const float* c1v_b = (const float*)d_in[12];
  const float* c2v_w = (const float*)d_in[13];
  const float* c2v_b = (const float*)d_in[14];
  const float* c3v_w = (const float*)d_in[15];
  const float* c3v_b = (const float*)d_in[16];
  // BN params: bn1q(17-20) bn2q(21-24) bn1k(25-28) bn2k(29-32) bn1v(33-36) bn2v(37-40)
  const float* co_w  = (const float*)d_in[41];
  const float* co_b  = (const float*)d_in[42];

  char* wp = (char*)d_ws;
  auto carve = [&](size_t bytes) {
    void* p = (void*)wp;
    wp += (bytes + 255) & ~(size_t)255;
    return p;
  };
  float*  h   = (float*)carve((size_t)NB * KD * MD * 4);
  float*  Q1  = (float*)carve((size_t)NB * 64 * MD * 4);
  float*  K1  = (float*)carve((size_t)NB * 64 * MD * 4);
  float*  V1  = (float*)carve((size_t)NB * 64 * MD * 4);
  bf16_t* Q2t = (bf16_t*)carve((size_t)NB * MD * KD * 2);
  bf16_t* K2t = (bf16_t*)carve((size_t)NB * MD * KD * 2);
  bf16_t* V2t = (bf16_t*)carve((size_t)NB * MD * KD * 2);
  bf16_t* W3q = (bf16_t*)carve((size_t)NN * KD * 2);
  bf16_t* W3k = (bf16_t*)carve((size_t)NN * KD * 2);
  bf16_t* W3v = (bf16_t*)carve((size_t)NN * KD * 2);
  bf16_t* Qb  = (bf16_t*)carve((size_t)NB * NN * MD * 2);
  bf16_t* Kb  = (bf16_t*)carve((size_t)NB * NN * MD * 2);
  bf16_t* Vb  = (bf16_t*)carve((size_t)NB * NN * MD * 2);

  k_fc1<<<4096, 256, 0, stream>>>(x, fc1_w, fc1_b, h);

  k_layer1<<<512, 256, 0, stream>>>(
      h, c1q_w, c1q_b, c1v_w, c1v_b,
      (const float*)d_in[17], (const float*)d_in[18], (const float*)d_in[19], (const float*)d_in[20],
      (const float*)d_in[25], (const float*)d_in[26], (const float*)d_in[27], (const float*)d_in[28],
      (const float*)d_in[33], (const float*)d_in[34], (const float*)d_in[35], (const float*)d_in[36],
      Q1, K1, V1);

  k_layer2<<<1024, 256, 0, stream>>>(
      Q1, K1, V1, c2q_w, c2q_b, c2v_w, c2v_b,
      (const float*)d_in[21], (const float*)d_in[22], (const float*)d_in[23], (const float*)d_in[24],
      (const float*)d_in[29], (const float*)d_in[30], (const float*)d_in[31], (const float*)d_in[32],
      (const float*)d_in[37], (const float*)d_in[38], (const float*)d_in[39], (const float*)d_in[40],
      Q2t, K2t, V2t);

  k_cvtw<<<(NN * KD) / 256, 256, 0, stream>>>(c3q_w, c3k_w, c3v_w, W3q, W3k, W3v);

  dim3 g3(32, 8);
  const float qscale = 0.08838834764831845f;  // 1/sqrt(K_DIM=128)
  k_layer3<<<g3, 256, 0, stream>>>(W3q, c3q_b, Q2t, Qb, qscale);
  k_layer3<<<g3, 256, 0, stream>>>(W3k, c3k_b, K2t, Kb, 1.0f);
  k_layer3<<<g3, 256, 0, stream>>>(W3v, c3v_b, V2t, Vb, 1.0f);

  k_attn<<<g3, 256, 0, stream>>>(Qb, Kb, Vb, co_w, co_b, (float*)d_out);
}
